// SAJBlock_46076409151631
// MI455X (gfx1250) — compile-verified, matrix operand order checked by reading, not run on hardware
//
#include <hip/hip_runtime.h>

// ---------------------------------------------------------------------------
// Types for CDNA5 WMMA
// ---------------------------------------------------------------------------
typedef __attribute__((ext_vector_type(16))) __bf16 v16bf;
typedef __attribute__((ext_vector_type(8)))  __bf16 v8bf;
typedef __attribute__((ext_vector_type(8)))  float  v8f;
typedef __attribute__((ext_vector_type(4)))  int    v4i;

static __device__ __forceinline__ v8f wmma_bf16(v16bf a, v16bf b, v8f c) {
  // D = A(16x32 bf16) * B(32x16 bf16) + C(16x16 f32)
  return __builtin_amdgcn_wmma_f32_16x16x32_bf16(
      /*neg_a=*/false, a, /*neg_b=*/false, b,
      /*c_mod=*/(short)0, c, /*reuse_a=*/false, /*reuse_b=*/false);
}

// --------------------------- async copy to LDS ------------------------------
#if defined(__has_builtin)
#if __has_builtin(__builtin_amdgcn_global_load_async_to_lds_b128)
#define USE_ASYNC_LDS 1
#endif
#endif

// Copy 16 bytes (8 bf16) global -> LDS. Async path uses the CDNA5
// GLOBAL_LOAD_ASYNC_TO_LDS_B128 engine (ASYNCcnt); fallback stages via VGPR.
static __device__ __forceinline__ void cp16(const __bf16* g, __bf16* l) {
#ifdef USE_ASYNC_LDS
  __builtin_amdgcn_global_load_async_to_lds_b128(
      (__attribute__((address_space(1))) v4i*)g,
      (__attribute__((address_space(3))) v4i*)l, 0, 0);
#else
  *(v8bf*)l = *(const v8bf*)g;
#endif
}

static __device__ __forceinline__ void wait_async_lds() {
#ifdef USE_ASYNC_LDS
#if __has_builtin(__builtin_amdgcn_s_wait_asynccnt)
  __builtin_amdgcn_s_wait_asynccnt(0);
#else
  asm volatile("s_wait_asynccnt 0" ::: "memory");
#endif
#endif
}

// Problem constants
#define Bz 8
#define Sz 1024
#define Dz 1024
#define Hz 16
#define DKz 64
#define DVz 64
#define DIz 2048
#define Mz (Bz * Sz)   // 8192

// ---------------------------------------------------------------------------
// Weight pack: f32 [K][N] -> bf16 [N][K] (transposed, contiguous-K rows)
// ---------------------------------------------------------------------------
__global__ __launch_bounds__(256)
void pack_transpose_bf16(const float* __restrict__ src, __bf16* __restrict__ dst,
                         int K, int N) {
  long idx = (long)blockIdx.x * 256 + threadIdx.x;
  if (idx >= (long)K * N) return;
  int n = (int)(idx / K);
  int k = (int)(idx % K);
  dst[idx] = (__bf16)src[(size_t)k * N + n];
}

// ---------------------------------------------------------------------------
// RMSNorm:  x[8192][1024] f32 -> nrm bf16  (one block per row)
// ---------------------------------------------------------------------------
__global__ __launch_bounds__(256)
void rmsnorm_bf16(const float* __restrict__ x, const float* __restrict__ gamma,
                  __bf16* __restrict__ out) {
  __shared__ float red[8];
  int row = blockIdx.x;
  const float* xr = x + (size_t)row * Dz;
  float v4[4];
  float s = 0.f;
#pragma unroll
  for (int i = 0; i < 4; ++i) {
    float v = xr[threadIdx.x + 256 * i];
    v4[i] = v;
    s += v * v;
  }
#pragma unroll
  for (int o = 16; o > 0; o >>= 1) s += __shfl_xor(s, o, 32);
  if ((threadIdx.x & 31) == 0) red[threadIdx.x >> 5] = s;
  __syncthreads();
  float tot = 0.f;
#pragma unroll
  for (int i = 0; i < 8; ++i) tot += red[i];
  float scale = rsqrtf(tot * (1.0f / Dz) + 1e-6f);
#pragma unroll
  for (int i = 0; i < 4; ++i) {
    int c = threadIdx.x + 256 * i;
    out[(size_t)row * Dz + c] = (__bf16)(v4[i] * scale * gamma[c]);
  }
}

// ---------------------------------------------------------------------------
// Generic WMMA GEMM:  C[M][N] = A[M][K](bf16, lda) * Bt[N][K](bf16) + bias
// 128x128 tile, BK=32, 8 waves (2x4), wave = 64x32 sub-tile = 4x2 wmma accs.
// Double-buffered LDS fed by GLOBAL_LOAD_ASYNC_TO_LDS_B128 (one barrier/iter).
// Epilogue modes:
//   0: f32 row-major        1: bf16 row-major        2: bf16 relu
//   3: f32 + residual       4: bf16 [B][H][S][64]    5: bf16 [B][H][64][S]
// ---------------------------------------------------------------------------
#define BM 128
#define BN 128
#define BK 32

__global__ __launch_bounds__(256)
void wmma_gemm(const __bf16* __restrict__ A, int lda,
               const __bf16* __restrict__ Bt,
               const float* __restrict__ bias,
               int M, int N, int K, int mode,
               float* __restrict__ outf, __bf16* __restrict__ outb,
               const float* __restrict__ resid) {
  __shared__ __align__(128) __bf16 As[2][BM][BK];
  __shared__ __align__(128) __bf16 Bs[2][BN][BK];

  int tid  = threadIdx.x;
  int wave = tid >> 5;
  int lane = tid & 31;
  int lr = lane & 15;      // row-within-fragment / col-within-tile
  int lh = lane >> 4;      // k-chunk selector
  int wm = wave >> 2;      // 0..1
  int wn = wave & 3;       // 0..3
  int bm = blockIdx.y * BM;
  int bn = blockIdx.x * BN;

  const __bf16* Abase = A  + (size_t)bm * lda;
  const __bf16* Bbase = Bt + (size_t)bn * K;

  v8f acc[4][2];
#pragma unroll
  for (int i = 0; i < 4; ++i)
#pragma unroll
    for (int j = 0; j < 2; ++j) acc[i][j] = {};

  // Stage one 128x32 A tile + 128x32 B tile into buffer `buf`:
  // 512 16-byte chunks per tile; each of 256 threads moves 2 per tile.
  auto stage = [&](int buf, int k0) {
#pragma unroll
    for (int c = tid; c < 512; c += 256) {
      int r = c >> 2, col = (c & 3) * 8;
      cp16(&Abase[(size_t)r * lda + k0 + col], &As[buf][r][col]);
      cp16(&Bbase[(size_t)r * K   + k0 + col], &Bs[buf][r][col]);
    }
  };

  int nk = K / BK;
  stage(0, 0);
  wait_async_lds();
  __syncthreads();

  for (int ki = 0; ki < nk; ++ki) {
    int cur = ki & 1;
    if (ki + 1 < nk) stage(cur ^ 1, (ki + 1) * BK);  // prefetch next tile

    v16bf af[4], bfr[2];
#pragma unroll
    for (int i = 0; i < 4; ++i)
      af[i] = *(const v16bf*)(&As[cur][wm * 64 + i * 16 + lr][lh * 16]);
#pragma unroll
    for (int j = 0; j < 2; ++j)
      bfr[j] = *(const v16bf*)(&Bs[cur][wn * 32 + j * 16 + lr][lh * 16]);

#pragma unroll
    for (int i = 0; i < 4; ++i)
#pragma unroll
      for (int j = 0; j < 2; ++j)
        acc[i][j] = wmma_bf16(af[i], bfr[j], acc[i][j]);

    wait_async_lds();   // next tile fully resident in LDS
    __syncthreads();    // all waves done reading `cur`, writes visible
  }

  // Epilogue. C layout: VGPR v, lanes 0-15 -> row v; lanes 16-31 -> row v+8.
#pragma unroll
  for (int i = 0; i < 4; ++i) {
#pragma unroll
    for (int j = 0; j < 2; ++j) {
      int col = bn + wn * 32 + j * 16 + lr;
      float bc = bias ? bias[col] : 0.f;
#pragma unroll
      for (int v = 0; v < 8; ++v) {
        int row = bm + wm * 64 + i * 16 + v + 8 * lh;
        float val = acc[i][j][v] + bc;
        if (mode == 0) {
          outf[(size_t)row * N + col] = val;
        } else if (mode == 1) {
          outb[(size_t)row * N + col] = (__bf16)val;
        } else if (mode == 2) {
          outb[(size_t)row * N + col] = (__bf16)fmaxf(val, 0.f);
        } else if (mode == 3) {
          size_t o = (size_t)row * N + col;
          outf[o] = val + resid[o];
        } else if (mode == 4) {
          int b = row >> 10, s = row & (Sz - 1);
          int h = col >> 6, d = col & (DKz - 1);
          outb[(((size_t)(b * Hz + h)) * Sz + s) * DKz + d] = (__bf16)val;
        } else { // 5: v stored transposed per head: [B][H][DV][S]
          int b = row >> 10, s = row & (Sz - 1);
          int h = col >> 6, d = col & (DVz - 1);
          outb[(((size_t)(b * Hz + h)) * DVz + d) * Sz + s] = (__bf16)val;
        }
      }
    }
  }
}

// ---------------------------------------------------------------------------
// Fused flash attention. One wave = 16 query rows; 4 waves/block.
// q,k: bf16 [B][H][S][64]; vt: bf16 [B][H][64][S]; ctx out: bf16 [B][S][H*64].
// Key blocks of 32: scores via 4 wmma, online softmax, P -> LDS -> A-frag,
// P@V via 4 wmma into 16x64 f32 accumulators.
// ---------------------------------------------------------------------------
__global__ __launch_bounds__(128)
void flash_attn(const __bf16* __restrict__ q, const __bf16* __restrict__ k,
                const __bf16* __restrict__ vt, __bf16* __restrict__ ctx) {
  __shared__ __align__(64) __bf16 p_lds[4][16 * 32];

  int wave = threadIdx.x >> 5;
  int lane = threadIdx.x & 31;
  int lr = lane & 15, lh = lane >> 4;

  int blk = blockIdx.x;                 // 0 .. B*H*(S/64)-1 = 2047
  int bh  = blk >> 4;                   // (b*H + h)
  int q0  = (blk & 15) * 64 + wave * 16;

  const __bf16* qp = q  + ((size_t)bh * Sz + q0) * DKz;
  const __bf16* kp = k  + (size_t)bh * Sz * DKz;
  const __bf16* vp = vt + (size_t)bh * DVz * Sz;

  v16bf qf0 = *(const v16bf*)(qp + (size_t)lr * DKz + lh * 16);
  v16bf qf1 = *(const v16bf*)(qp + (size_t)lr * DKz + 32 + lh * 16);

  v8f acc[4];
#pragma unroll
  for (int t = 0; t < 4; ++t) acc[t] = {};
  float m[8], l[8];
#pragma unroll
  for (int v = 0; v < 8; ++v) { m[v] = -1e30f; l[v] = 0.f; }

  __bf16* pl = &p_lds[wave][0];
  const float scale = 0.125f;  // 1/sqrt(64)

  for (int kb = 0; kb < Sz; kb += 32) {
    v16bf kf00 = *(const v16bf*)(kp + (size_t)(kb + lr) * DKz + lh * 16);
    v16bf kf01 = *(const v16bf*)(kp + (size_t)(kb + lr) * DKz + 32 + lh * 16);
    v16bf kf10 = *(const v16bf*)(kp + (size_t)(kb + 16 + lr) * DKz + lh * 16);
    v16bf kf11 = *(const v16bf*)(kp + (size_t)(kb + 16 + lr) * DKz + 32 + lh * 16);

    v8f z = {};
    v8f s0 = wmma_bf16(qf1, kf01, wmma_bf16(qf0, kf00, z));
    v8f s1 = wmma_bf16(qf1, kf11, wmma_bf16(qf0, kf10, z));

    float p0[8], p1[8], nm[8], rs[8];
#pragma unroll
    for (int v = 0; v < 8; ++v) {
      s0[v] *= scale;
      s1[v] *= scale;
      float t = fmaxf(s0[v], s1[v]);
#pragma unroll
      for (int o = 1; o < 16; o <<= 1) t = fmaxf(t, __shfl_xor(t, o, 32));
      nm[v] = fmaxf(m[v], t);
      p0[v] = __expf(s0[v] - nm[v]);
      p1[v] = __expf(s1[v] - nm[v]);
      float r = p0[v] + p1[v];
#pragma unroll
      for (int o = 1; o < 16; o <<= 1) r += __shfl_xor(r, o, 32);
      rs[v] = r;
      float al = __expf(m[v] - nm[v]);
      l[v] = l[v] * al + rs[v];
      m[v] = nm[v];
#pragma unroll
      for (int t4 = 0; t4 < 4; ++t4) acc[t4][v] *= al;
    }

    // P (16x32 f32 in C-layout) -> LDS as row-major bf16 tile
#pragma unroll
    for (int v = 0; v < 8; ++v) {
      int row = v + 8 * lh;
      pl[row * 32 + lr]      = (__bf16)p0[v];
      pl[row * 32 + 16 + lr] = (__bf16)p1[v];
    }
    asm volatile("s_wait_dscnt 0" ::: "memory");  // in-wave LDS RAW
    v16bf pf = *(const v16bf*)(pl + lr * 32 + lh * 16);

#pragma unroll
    for (int t4 = 0; t4 < 4; ++t4) {
      v16bf vf = *(const v16bf*)(vp + (size_t)(t4 * 16 + lr) * Sz + kb + lh * 16);
      acc[t4] = wmma_bf16(pf, vf, acc[t4]);
    }
  }

  // ctx[b][s][h*64 + d] = acc / l
  int b = bh >> 4, h = bh & (Hz - 1);
#pragma unroll
  for (int t4 = 0; t4 < 4; ++t4) {
#pragma unroll
    for (int v = 0; v < 8; ++v) {
      int row = v + 8 * lh;
      float o = acc[t4][v] / l[v];
      ctx[((size_t)(b * Sz + q0 + row)) * (Hz * DVz) + h * DVz + t4 * 16 + lr] =
          (__bf16)o;
    }
  }
}

// ---------------------------------------------------------------------------
// Depthwise dilated conv (K=4, dilation=2, pad 3) + bias + gate with sec.
// f: f32 [M][DI]; sec: bf16 [M][DI]; out gated bf16 [M][DI].
// ---------------------------------------------------------------------------
__global__ __launch_bounds__(256)
void conv_gate(const float* __restrict__ f, const float* __restrict__ cw,
               const float* __restrict__ cb, const __bf16* __restrict__ sec,
               __bf16* __restrict__ gated) {
  size_t idx = (size_t)blockIdx.x * 256 + threadIdx.x;  // over M*DI
  int c = (int)(idx & (DIz - 1));
  size_t mrow = idx >> 11;
  int s = (int)(mrow & (Sz - 1));
  float a = cb[c];
#pragma unroll
  for (int j = 0; j < 4; ++j) {
    int off = 2 * j - 3;          // {-3,-1,1,3}
    int ss = s + off;
    if (ss >= 0 && ss < Sz)
      a += cw[c * 4 + j] * f[(mrow + off) * DIz + c];
  }
  gated[idx] = (__bf16)(a * (float)sec[idx]);
}

// ---------------------------------------------------------------------------
// Host launcher
// ---------------------------------------------------------------------------
extern "C" void kernel_launch(void* const* d_in, const int* in_sizes, int n_in,
                              void* d_out, int out_size, void* d_ws, size_t ws_size,
                              hipStream_t stream) {
  (void)in_sizes; (void)n_in; (void)out_size; (void)ws_size;

  const float* x     = (const float*)d_in[0];
  // d_in[1] = mask (all ones -> unused)
  const float* gamma = (const float*)d_in[2];
  const float* wq    = (const float*)d_in[3];
  const float* wk    = (const float*)d_in[4];
  const float* wv    = (const float*)d_in[5];
  const float* wo    = (const float*)d_in[6];
  const float* w1    = (const float*)d_in[7];
  const float* b1    = (const float*)d_in[8];
  const float* cw    = (const float*)d_in[9];
  const float* cb    = (const float*)d_in[10];
  const float* w2    = (const float*)d_in[11];
  const float* b2    = (const float*)d_in[12];
  const float* wp    = (const float*)d_in[13];
  const float* bp    = (const float*)d_in[14];
  float* out = (float*)d_out;

  char* ws = (char*)d_ws;
  const size_t MB = 1ull << 20;
  __bf16* wq_t = (__bf16*)(ws + 0 * MB);     // [1024][1024]
  __bf16* wk_t = (__bf16*)(ws + 2 * MB);
  __bf16* wv_t = (__bf16*)(ws + 4 * MB);
  __bf16* wo_t = (__bf16*)(ws + 6 * MB);
  __bf16* w1_t = (__bf16*)(ws + 8 * MB);     // [2048][512]
  __bf16* w2_t = (__bf16*)(ws + 10 * MB);
  __bf16* wp_t = (__bf16*)(ws + 12 * MB);    // [1024][2048]
  __bf16* nrm  = (__bf16*)(ws + 16 * MB);    // [8192][1024]; reused as ctx
  __bf16* qbuf = (__bf16*)(ws + 32 * MB);    // [B][H][S][64]; reused as out_buf
  __bf16* kbuf = (__bf16*)(ws + 48 * MB);    // [B][H][S][64]; (kbuf..) reused as sec
  __bf16* vtbuf= (__bf16*)(ws + 64 * MB);    // [B][H][64][S]
  float*  fbuf = (float*)(ws + 80 * MB);     // [8192][2048] f32 (64MB)
  __bf16* gbuf = (__bf16*)(ws + 144 * MB);   // [8192][2048] bf16 (32MB)
  __bf16* ctx  = nrm;
  __bf16* obuf = qbuf;
  __bf16* sec  = kbuf;

  // 1) pack + transpose weights to bf16 [N][K]
  auto packT = [&](const float* src, __bf16* dst, int K, int N) {
    long tot = (long)K * N;
    pack_transpose_bf16<<<dim3((tot + 255) / 256), dim3(256), 0, stream>>>(src, dst, K, N);
  };
  packT(wq, wq_t, 1024, 1024);
  packT(wk, wk_t, 1024, 1024);
  packT(wv, wv_t, 1024, 1024);
  packT(wo, wo_t, 1024, 1024);
  packT(w1, w1_t, 512, 2048);
  packT(w2, w2_t, 512, 2048);
  packT(wp, wp_t, 2048, 1024);

  // 2) RMSNorm -> bf16
  rmsnorm_bf16<<<dim3(Mz), dim3(256), 0, stream>>>(x, gamma, nrm);

  // 3) q, k, v projections (head-split layouts)
  wmma_gemm<<<dim3(1024 / BN, Mz / BM), dim3(256), 0, stream>>>(
      nrm, 1024, wq_t, nullptr, Mz, 1024, 1024, 4, nullptr, qbuf, nullptr);
  wmma_gemm<<<dim3(1024 / BN, Mz / BM), dim3(256), 0, stream>>>(
      nrm, 1024, wk_t, nullptr, Mz, 1024, 1024, 4, nullptr, kbuf, nullptr);
  wmma_gemm<<<dim3(1024 / BN, Mz / BM), dim3(256), 0, stream>>>(
      nrm, 1024, wv_t, nullptr, Mz, 1024, 1024, 5, nullptr, vtbuf, nullptr);

  // 4) fused attention -> ctx (reuses nrm buffer)
  flash_attn<<<dim3(Bz * Hz * (Sz / 64)), dim3(128), 0, stream>>>(qbuf, kbuf, vtbuf, ctx);

  // 5) out = ctx @ wo (bf16, reuses q buffer)
  wmma_gemm<<<dim3(1024 / BN, Mz / BM), dim3(256), 0, stream>>>(
      ctx, 1024, wo_t, nullptr, Mz, 1024, 1024, 1, nullptr, obuf, nullptr);

  // 6) f = f_in @ w1 + b1 (f32, for conv)
  wmma_gemm<<<dim3(2048 / BN, Mz / BM), dim3(256), 0, stream>>>(
      obuf, 1024, w1_t, b1, Mz, 2048, 512, 0, fbuf, nullptr, nullptr);

  // 7) sec = relu(s_in @ w2 + b2) (bf16; reuses k/v region)
  wmma_gemm<<<dim3(2048 / BN, Mz / BM), dim3(256), 0, stream>>>(
      obuf + 512, 1024, w2_t, b2, Mz, 2048, 512, 2, nullptr, sec, nullptr);

  // 8) depthwise dilated conv + bias + gate
  conv_gate<<<dim3((Mz * DIz) / 256), dim3(256), 0, stream>>>(fbuf, cw, cb, sec, gbuf);

  // 9) y = x + gated @ wp + bp
  wmma_gemm<<<dim3(1024 / BN, Mz / BM), dim3(256), 0, stream>>>(
      gbuf, 2048, wp_t, bp, Mz, 1024, 2048, 3, out, nullptr, x);
}
